// LSTMAutoencoder_15058155340177
// MI455X (gfx1250) — matrix operand order repaired and checked by reading, not compiled, not run on hardware
//
#include <hip/hip_runtime.h>

typedef __attribute__((ext_vector_type(16))) __bf16 v16bf;
typedef __attribute__((ext_vector_type(8)))  __bf16 v8bf;
typedef __attribute__((ext_vector_type(8)))  float  v8f;

#define Bc   64
#define Tc   128
#define Ic   256
#define Hc   1024
#define G4   4096      // 4*H
#define K0c  1280      // I + H   (layer-0 concat K)
#define K1c  2048      // H + H   (layer-1 concat K)
#define NWG  32
#define TPB  256
#define NLANES (NWG*TPB)   // 8192

// ---------------------------------------------------------------- helpers ---

__device__ __forceinline__ v16bf ld_frag(const __bf16* p0, const __bf16* p1) {
  v8bf lo = *(const v8bf*)p0;
  v8bf hi = *(const v8bf*)p1;
  return __builtin_shufflevector(lo, hi, 0,1,2,3,4,5,6,7,8,9,10,11,12,13,14,15);
}
// A-matrix 16x32 bf16 fragment (ISA 7.12.2): row m=lane&15; half=lane>>4 holds
// K = half*8 + [0..7]  and  half*8 + 16 + [0..7]
__device__ __forceinline__ v16bf fragA(const __bf16* p) { return ld_frag(p, p + 16); }
// B-matrix 32x16 bf16 fragment: col n=lane&15; half=lane>>4 holds K = half*8 + [0..7] and half*8 + 16 + [0..7]
__device__ __forceinline__ v16bf fragB(const __bf16* p) { return ld_frag(p, p + 16); }

__device__ __forceinline__ v8f wmma_bf16(v16bf a, v16bf b, v8f c) {
  return __builtin_amdgcn_wmma_f32_16x16x32_bf16(false, a, false, b, (short)0, c, false, false);
}

__device__ __forceinline__ float sigf(float x) { return 1.0f / (1.0f + __expf(-x)); }

// device-wide barrier: bar[0] = arrival counter, bar[1] = generation
__device__ __forceinline__ void grid_sync(unsigned* bar) {
  __syncthreads();
  if (threadIdx.x == 0) {
    __threadfence();   // release all prior stores to device scope
    unsigned g = __hip_atomic_load(&bar[1], __ATOMIC_RELAXED, __HIP_MEMORY_SCOPE_AGENT);
    unsigned a = __hip_atomic_fetch_add(&bar[0], 1u, __ATOMIC_ACQ_REL, __HIP_MEMORY_SCOPE_AGENT);
    if (a == NWG - 1) {
      __hip_atomic_store(&bar[0], 0u, __ATOMIC_RELAXED, __HIP_MEMORY_SCOPE_AGENT);
      __hip_atomic_store(&bar[1], g + 1u, __ATOMIC_RELEASE, __HIP_MEMORY_SCOPE_AGENT);
    } else {
      while (__hip_atomic_load(&bar[1], __ATOMIC_ACQUIRE, __HIP_MEMORY_SCOPE_AGENT) == g)
        __builtin_amdgcn_s_sleep(2);
    }
    __threadfence();   // acquire side
  }
  __syncthreads();
}

// gates(64 x 4096) = A(64 x Kc) @ W(4096 x Kc)^T   -> fp32 scratch G
// Each of 256 waves owns ONE 16-wide N-tile for all of M=64, so every weight
// element is loaded exactly once device-wide. A (<=256KB bf16) is staged into
// LDS per workgroup and A fragments come from ds_load_b128.
__device__ void gemm_gates(const __bf16* __restrict__ Asrc, __bf16* __restrict__ Alds,
                           const __bf16* __restrict__ W, int Kc,
                           float* __restrict__ Gm, int wave, int lane)
{
  // ---- stage A (64 x Kc bf16) global -> LDS, whole workgroup cooperates ----
  {
    v8bf*       d = (v8bf*)Alds;
    const v8bf* s = (const v8bf*)Asrc;
    const int chunks = (Bc * Kc) >> 3;            // 16B chunks
    for (int i = threadIdx.x; i < chunks; i += TPB) d[i] = s[i];
  }
  __syncthreads();

  const int nbase = wave << 4;                    // 256 waves x 16 cols = 4096
  const int half  = lane >> 4;
  const int l15   = lane & 15;
  const __bf16* wp = W    + (size_t)(nbase + l15) * Kc + (half << 3);
  const __bf16* ap = Alds + l15 * Kc + (half << 3);
  const int arow16 = Kc << 4;                     // 16 A-rows
  v8f acc0 = {0,0,0,0,0,0,0,0}, acc1 = acc0, acc2 = acc0, acc3 = acc0;
  for (int k = 0; k < Kc; k += 32) {
    __builtin_prefetch(wp + k + 64, 0, 3);        // global_prefetch_b8 (weights in L2)
    v16bf bf = fragB(wp + k);                     // 2x global_load_b128 (unique rows)
    v16bf a0 = fragA(ap + k);                     // 8x ds_load_b128 total
    v16bf a1 = fragA(ap + arow16 + k);
    v16bf a2 = fragA(ap + 2*arow16 + k);
    v16bf a3 = fragA(ap + 3*arow16 + k);
    acc0 = wmma_bf16(a0, bf, acc0);
    acc1 = wmma_bf16(a1, bf, acc1);
    acc2 = wmma_bf16(a2, bf, acc2);
    acc3 = wmma_bf16(a3, bf, acc3);
  }
  // C/D layout: VGPR r -> M = half*8 + r, N = l15
  float* g0 = Gm + (size_t)(half << 3) * G4 + nbase + l15;
  #pragma unroll
  for (int r = 0; r < 8; ++r) {
    g0[(size_t)(r     ) * G4] = acc0[r];
    g0[(size_t)(r + 16) * G4] = acc1[r];
    g0[(size_t)(r + 32) * G4] = acc2[r];
    g0[(size_t)(r + 48) * G4] = acc3[r];
  }
}

// decoder projection: out(64 x 256) = h1(64 x 1024) @ Wout^T + bout
// (small; 64 waves, global-direct reads, no __syncthreads inside)
__device__ void gemm_proj(const __bf16* __restrict__ A,   // A1 + 1024 (stride 2048)
                          const __bf16* __restrict__ Wo,  // 256 x 1024 bf16
                          const float* __restrict__ bout,
                          float* __restrict__ out, int t,
                          __bf16* __restrict__ fb,        // A0 feedback (stride 1280)
                          int wave, int lane)
{
  if (wave >= 64) return;
  const int mt = wave & 3, nt = wave >> 2;
  const int mbase = mt << 4, nbase = nt << 4;
  const int half = lane >> 4, l15 = lane & 15;
  const __bf16* ap = A  + (mbase + l15) * K1c + (half << 3);
  const __bf16* wp = Wo + (nbase + l15) * Hc  + (half << 3);
  v8f acc = {0,0,0,0,0,0,0,0};
  for (int k = 0; k < Hc; k += 32)
    acc = wmma_bf16(fragA(ap + k), fragB(wp + k), acc);
  const int n = nbase + l15;
  const float bias = bout[n];
  #pragma unroll
  for (int r = 0; r < 8; ++r) {
    const int m = mbase + (half << 3) + r;
    const float v = acc[r] + bias;
    out[(size_t)m * Tc * Ic + (size_t)t * Ic + n] = v;   // (B,T,I)
    fb[m * K0c + n] = (__bf16)v;                          // next-step decoder input
  }
}

// elementwise LSTM cell update over 64x1024: bias add, gate nonlinearities,
// cell-state update, write bf16 h into concat activation buffer(s).
__device__ void lstm_elem(const float* __restrict__ Gm, const float* __restrict__ bias,
                          float* __restrict__ c,
                          __bf16* __restrict__ h1dst, int s1,
                          __bf16* __restrict__ h2dst, int s2, int tid)
{
  #pragma unroll
  for (int e = 0; e < (Bc*Hc)/NLANES; ++e) {          // 8 elements / lane
    const int idx = tid + e * NLANES;
    const int b = idx >> 10, u = idx & (Hc - 1);
    const float* gr = Gm + (size_t)b * G4;
    const float gi = gr[u]          + bias[u];
    const float gf = gr[Hc + u]     + bias[Hc + u];
    const float gg = gr[2*Hc + u]   + bias[2*Hc + u];
    const float go = gr[3*Hc + u]   + bias[3*Hc + u];
    const float cn = sigf(gf) * c[idx] + sigf(gi) * tanhf(gg);
    c[idx] = cn;
    const __bf16 hb = (__bf16)(sigf(go) * tanhf(cn));
    h1dst[b * s1 + u] = hb;
    if (h2dst) h2dst[b * s2 + u] = hb;
  }
}

__device__ void fill_x(__bf16* __restrict__ A0, const float* __restrict__ x, int t, int tid) {
  #pragma unroll
  for (int e = 0; e < (Bc*Ic)/NLANES; ++e) {          // 2 elements / lane
    const int idx = tid + e * NLANES;
    const int b = idx >> 8, k = idx & (Ic - 1);
    A0[b * K0c + k] = (__bf16)x[((size_t)b * Tc + t) * Ic + k];
  }
}

// ---------------------------------------------------------------- kernels ---

// dst(rows x (Ka+Kb)) bf16 = concat(srcA(rows x Ka), srcB(rows x Kb)) fp32
__global__ void concat_convert(const float* __restrict__ srcA, int Ka,
                               const float* __restrict__ srcB, int Kb,
                               __bf16* __restrict__ dst, int rows)
{
  const int K = Ka + Kb;
  const long long total = (long long)rows * K;
  for (long long i = (long long)blockIdx.x * blockDim.x + threadIdx.x;
       i < total; i += (long long)gridDim.x * blockDim.x) {
    const int r = (int)(i / K), k = (int)(i % K);
    const float v = (k < Ka) ? srcA[(size_t)r * Ka + k] : srcB[(size_t)r * Kb + (k - Ka)];
    dst[i] = (__bf16)v;
  }
}

__global__ void init_state(__bf16* A0, __bf16* A1, float* c0, float* c1, unsigned* bar)
{
  const int tid = blockIdx.x * blockDim.x + threadIdx.x;
  const int nt  = gridDim.x * blockDim.x;
  for (int i = tid; i < Bc*K0c; i += nt) A0[i] = (__bf16)0.0f;
  for (int i = tid; i < Bc*K1c; i += nt) A1[i] = (__bf16)0.0f;
  for (int i = tid; i < Bc*Hc;  i += nt) { c0[i] = 0.0f; c1[i] = 0.0f; }
  if (tid == 0) { bar[0] = 0u; bar[1] = 0u; }
}

__global__ __launch_bounds__(TPB) void lstm_persistent(
    const float* __restrict__ x,
    const float* __restrict__ b0e, const float* __restrict__ b1e,
    const float* __restrict__ b0d, const float* __restrict__ b1d,
    const float* __restrict__ bout,
    const __bf16* __restrict__ W0e, const __bf16* __restrict__ W1e,
    const __bf16* __restrict__ W0d, const __bf16* __restrict__ W1d,
    const __bf16* __restrict__ Wo,
    __bf16* __restrict__ A0, __bf16* __restrict__ A1,
    float* __restrict__ c0, float* __restrict__ c1,
    float* __restrict__ Gm, unsigned* __restrict__ bar,
    float* __restrict__ out)
{
  // 64 x 2048 bf16 = 256KB of CDNA5's 320KB LDS (one workgroup per WGP)
  __shared__ __bf16 Alds[Bc * K1c];

  const int tid  = blockIdx.x * blockDim.x + threadIdx.x;
  const int wave = tid >> 5;
  const int lane = tid & 31;

  // stage 0: encoder input for t=0
  fill_x(A0, x, 0, tid);
  grid_sync(bar);

  // ---- encoder: 2 stacked LSTM layers over T steps ----
  for (int t = 0; t < Tc; ++t) {
    gemm_gates(A0, Alds, W0e, K0c, Gm, wave, lane);           grid_sync(bar);
    lstm_elem(Gm, b0e, c0, A0 + Ic, K0c, A1, K1c, tid);       grid_sync(bar);
    gemm_gates(A1, Alds, W1e, K1c, Gm, wave, lane);           grid_sync(bar);
    lstm_elem(Gm, b1e, c1, A1 + Hc, K1c, (__bf16*)0, 0, tid);
    // prepare next input: x[:,t+1,:]; after last step, seed decoder with x[:,0,:]
    fill_x(A0, x, (t + 1 < Tc) ? t + 1 : 0, tid);
    grid_sync(bar);
  }

  // ---- decoder: autoregressive, states carried over from encoder ----
  for (int t = 0; t < Tc; ++t) {
    gemm_gates(A0, Alds, W0d, K0c, Gm, wave, lane);           grid_sync(bar);
    lstm_elem(Gm, b0d, c0, A0 + Ic, K0c, A1, K1c, tid);       grid_sync(bar);
    gemm_gates(A1, Alds, W1d, K1c, Gm, wave, lane);           grid_sync(bar);
    lstm_elem(Gm, b1d, c1, A1 + Hc, K1c, (__bf16*)0, 0, tid); grid_sync(bar);
    gemm_proj(A1 + Hc, Wo, bout, out, t, A0, wave, lane);     grid_sync(bar);
  }
}

// ------------------------------------------------------------------ host ---

extern "C" void kernel_launch(void* const* d_in, const int* in_sizes, int n_in,
                              void* d_out, int out_size, void* d_ws, size_t ws_size,
                              hipStream_t stream) {
  (void)in_sizes; (void)n_in; (void)out_size; (void)ws_size;
  const float* x     = (const float*)d_in[0];
  const float* eWih0 = (const float*)d_in[1];
  const float* eWhh0 = (const float*)d_in[2];
  const float* eb0   = (const float*)d_in[3];
  const float* eWih1 = (const float*)d_in[4];
  const float* eWhh1 = (const float*)d_in[5];
  const float* eb1   = (const float*)d_in[6];
  const float* dWih0 = (const float*)d_in[7];
  const float* dWhh0 = (const float*)d_in[8];
  const float* db0   = (const float*)d_in[9];
  const float* dWih1 = (const float*)d_in[10];
  const float* dWhh1 = (const float*)d_in[11];
  const float* db1   = (const float*)d_in[12];
  const float* Wout  = (const float*)d_in[13];
  const float* bout  = (const float*)d_in[14];

  char* ws = (char*)d_ws;
  size_t off = 0;
  auto take = [&](size_t bytes) -> char* {
    char* p = ws + off;
    off = (off + bytes + 255) & ~(size_t)255;
    return p;
  };
  __bf16* W0e = (__bf16*)take((size_t)G4 * K0c * 2);
  __bf16* W1e = (__bf16*)take((size_t)G4 * K1c * 2);
  __bf16* W0d = (__bf16*)take((size_t)G4 * K0c * 2);
  __bf16* W1d = (__bf16*)take((size_t)G4 * K1c * 2);
  __bf16* Wo  = (__bf16*)take((size_t)Ic * Hc * 2);
  __bf16* A0  = (__bf16*)take((size_t)Bc * K0c * 2);
  __bf16* A1  = (__bf16*)take((size_t)Bc * K1c * 2);
  float*  c0  = (float*) take((size_t)Bc * Hc * 4);
  float*  c1  = (float*) take((size_t)Bc * Hc * 4);
  float*  Gm  = (float*) take((size_t)Bc * G4 * 4);
  unsigned* bar = (unsigned*)take(256);

  // one-time (per call) weight conversion: fp32 -> bf16, [Wih | Whh] concat
  concat_convert<<<512, TPB, 0, stream>>>(eWih0, Ic, eWhh0, Hc, W0e, G4);
  concat_convert<<<512, TPB, 0, stream>>>(eWih1, Hc, eWhh1, Hc, W1e, G4);
  concat_convert<<<512, TPB, 0, stream>>>(dWih0, Ic, dWhh0, Hc, W0d, G4);
  concat_convert<<<512, TPB, 0, stream>>>(dWih1, Hc, dWhh1, Hc, W1d, G4);
  concat_convert<<<64,  TPB, 0, stream>>>(Wout,  Hc, Wout,  0,  Wo,  Ic);
  init_state<<<64, TPB, 0, stream>>>(A0, A1, c0, c1, bar);

  lstm_persistent<<<NWG, TPB, 0, stream>>>(
      x, eb0, eb1, db0, db1, bout,
      W0e, W1e, W0d, W1d, Wo,
      A0, A1, c0, c1, Gm, bar, (float*)d_out);
}
